// Custom_Quantization_MultiHeadAttention_6725918785731
// MI455X (gfx1250) — matrix operand order repaired and checked by reading, not compiled, not run on hardware
//
#include <hip/hip_runtime.h>

#define BB   4
#define LL   2048
#define DD   512
#define KDIM 64
#define HH   8

typedef __attribute__((ext_vector_type(16))) _Float16 v16h;
typedef __attribute__((ext_vector_type(8)))  _Float16 v8h;
typedef __attribute__((ext_vector_type(8)))  float    v8f;

// ---------- WMMA helpers ----------

static __device__ __forceinline__ v8f wmma_f16(v16h a, v16h b, v8f c) {
    // v_wmma_f32_16x16x32_f16 : D = A(16x32 f16) * B(32x16 f16) + C(16x16 f32)
    return __builtin_amdgcn_wmma_f32_16x16x32_f16(
        /*neg_a=*/false, a, /*neg_b=*/false, b,
        /*c_mod=*/(short)0, c, /*reuse_a=*/false, /*reuse_b=*/false);
}

// A-fragment (16x32, row-major source): lane (g = lane>>4, m = lane&15) holds
// row m, k in [k0+g*16 .. +16) contiguous. Two b128 loads.
static __device__ __forceinline__ v16h ld_a16(const _Float16* p) {
    v8h lo = *(const v8h*)(p);
    v8h hi = *(const v8h*)(p + 8);
    return __builtin_shufflevector(lo, hi,
        0, 1, 2, 3, 4, 5, 6, 7, 8, 9, 10, 11, 12, 13, 14, 15);
}

// 16-lane-group reductions (wave32; xor masks < 16 stay inside the group)
static __device__ __forceinline__ float grp_max(float v) {
    v = fmaxf(v, __shfl_xor(v, 1));
    v = fmaxf(v, __shfl_xor(v, 2));
    v = fmaxf(v, __shfl_xor(v, 4));
    v = fmaxf(v, __shfl_xor(v, 8));
    return v;
}
static __device__ __forceinline__ float grp_sum(float v) {
    v += __shfl_xor(v, 1);
    v += __shfl_xor(v, 2);
    v += __shfl_xor(v, 4);
    v += __shfl_xor(v, 8);
    return v;
}

// ---------- Kernel: dequantize 512x512 int32 weight, store transposed f16 ----------
__global__ void __launch_bounds__(256)
dequant_transpose_k(const int* __restrict__ src, _Float16* __restrict__ dst,
                    const float* __restrict__ scale, const float* __restrict__ zp) {
    int idx = blockIdx.x * 256 + threadIdx.x;   // 0 .. 512*512-1, row-major (r,c)
    int r = idx >> 9;
    int c = idx & 511;
    float s = scale[0], z = zp[0];
    dst[c * 512 + r] = (_Float16)(((float)src[idx] - z) * s);
}

// ---------- Kernel: fp32 -> f16 convert ----------
__global__ void __launch_bounds__(256)
f32_to_f16_k(const float* __restrict__ x, _Float16* __restrict__ y, int n) {
    int i = blockIdx.x * 256 + threadIdx.x;
    if (i < n) y[i] = (_Float16)x[i];
}

// ---------- Kernel: GEMM (8192x512) = A(8192x512,f16) * Bt^T + bias ----------
// Bt is stored N-major (512 x 512): Bt[n][k] = B[k][n], so B-fragments are
// contiguous v16h loads. Each wave computes a 16x64 tile (4 accumulators).
// K-loop is register double-buffered: fragments for step k+1 are issued
// before the WMMAs of step k, so WMMAs wait on partial LOADcnt, not 0.
// mode 0: f16 out, (B,H,L,KD) layout (q,k)
// mode 1: f16 out, (B,H,KD,L) layout (v, transposed for the P*V stage)
// mode 2: f32 out, plain row-major (final output)
__global__ void __launch_bounds__(256)
gemm512_k(const _Float16* __restrict__ A, const _Float16* __restrict__ Bt,
          const float* __restrict__ bias, void* __restrict__ out, int mode) {
    int lane = threadIdx.x & 31;
    int wave = threadIdx.x >> 5;
    int g    = lane >> 4;
    int nn   = lane & 15;
    int row0 = blockIdx.x * 128 + wave * 16;
    int col0 = blockIdx.y * 64;

    v8f acc[4] = {};
    const _Float16* ap = A  + (size_t)(row0 + nn) * 512 + g * 16;
    const _Float16* bp = Bt + (size_t)(col0 + nn) * 512 + g * 16;

    v16h a_cur, a_nxt;
    v16h b_cur[4], b_nxt[4];
    a_cur = ld_a16(ap);
#pragma unroll
    for (int t = 0; t < 4; ++t)
        b_cur[t] = *(const v16h*)(bp + (size_t)t * 16 * 512);
    a_nxt = a_cur;
#pragma unroll
    for (int t = 0; t < 4; ++t) b_nxt[t] = b_cur[t];

#pragma unroll
    for (int kk = 0; kk < 16; ++kk) {
        if (kk < 15) {
            int k0n = (kk + 1) * 32;
            a_nxt = ld_a16(ap + k0n);
#pragma unroll
            for (int t = 0; t < 4; ++t)
                b_nxt[t] = *(const v16h*)(bp + (size_t)t * 16 * 512 + k0n);
        }
#pragma unroll
        for (int t = 0; t < 4; ++t)
            acc[t] = wmma_f16(a_cur, b_cur[t], acc[t]);
        a_cur = a_nxt;
#pragma unroll
        for (int t = 0; t < 4; ++t) b_cur[t] = b_nxt[t];
    }

#pragma unroll
    for (int t = 0; t < 4; ++t) {
        int col = col0 + t * 16 + nn;
        float bv = bias[col];
#pragma unroll
        for (int r = 0; r < 8; ++r) {
            int row = row0 + r + 8 * g;           // C layout: m = r + 8*(lane>>4)
            float val = acc[t][r] + bv;
            if (mode == 2) {
                ((float*)out)[(size_t)row * 512 + col] = val;
            } else {
                int b2 = row >> 11, i = row & 2047;   // row = b*L + i
                int kd = col >> 3,  h = col & 7;      // col = kd*H + h
                size_t dst = (mode == 0)
                    ? ((((size_t)b2 * HH + h) * LL + i) * KDIM + kd)
                    : ((((size_t)b2 * HH + h) * KDIM + kd) * LL + i);
                ((_Float16*)out)[dst] = (_Float16)val;
            }
        }
    }
}

// ---------- Kernel: flash attention per (b,h) ----------
// q_s,k_s: (B,H,L,KD) f16 row-major.  v_t: (B,H,KD,L) f16 (transposed).
// Block = 4 waves x 16 query rows = 64 rows. Loop over 64-key blocks:
//   S = Q*K^T (8 WMMAs), online softmax (v_exp_f32), P -> LDS (C->A layout
//   transpose), O += P*V (8 WMMAs). 1/sqrt(KD) folded into Q at preload.
// V fragments are loaded before the softmax so global latency overlaps the
// exp/shuffle VALU work; next block's K/V panels get global_prefetch_b8.
__global__ void __launch_bounds__(128)
attn_k(const _Float16* __restrict__ q_s, const _Float16* __restrict__ k_s,
       const _Float16* __restrict__ v_t, _Float16* __restrict__ o_s) {
    __shared__ __align__(16) _Float16 plds[4][16 * 64];

    int lane = threadIdx.x & 31;
    int wave = threadIdx.x >> 5;
    int g    = lane >> 4;
    int nn   = lane & 15;
    int bh   = blockIdx.x;           // 0..31
    int b    = bh >> 3, h = bh & 7;
    int qrow0 = blockIdx.y * 64 + wave * 16;

    const _Float16* qh = q_s + (size_t)bh * LL * KDIM;
    const _Float16* kh = k_s + (size_t)bh * LL * KDIM;
    const _Float16* vh = v_t + (size_t)bh * KDIM * LL;
    _Float16* pl = plds[wave];

    // Preload Q A-fragments, fold in softmax scale 1/sqrt(64) = 0.125
    v16h qf[2];
#pragma unroll
    for (int kc = 0; kc < 2; ++kc) {
        qf[kc] = ld_a16(qh + (size_t)(qrow0 + nn) * KDIM + kc * 32 + g * 16);
        qf[kc] = qf[kc] * (_Float16)0.125f;
    }

    float rmax[8], rsum[8];
    v8f oacc[4] = {};
#pragma unroll
    for (int r = 0; r < 8; ++r) { rmax[r] = -1e30f; rsum[r] = 0.f; }

    for (int kb = 0; kb < LL / 64; ++kb) {
        int key0 = kb * 64;

        // Prefetch next key block's K and V panels (one 128B line per lane)
        int nkey0 = key0 + 64;
        if (nkey0 < LL) {
            __builtin_prefetch(kh + (size_t)(nkey0 + lane * 2) * KDIM, 0, 3);
            __builtin_prefetch(vh + (size_t)(lane * 2) * LL + nkey0, 0, 3);
        }

        // Batch all K fragments, then the 8 S WMMAs
        v16h kf[8];
#pragma unroll
        for (int t = 0; t < 4; ++t)
#pragma unroll
            for (int kc = 0; kc < 2; ++kc)
                kf[t * 2 + kc] = *(const v16h*)(kh + (size_t)(key0 + t * 16 + nn) * KDIM
                                                   + kc * 32 + g * 16);
        v8f s[4];
#pragma unroll
        for (int t = 0; t < 4; ++t) {
            v8f st = {};
            st = wmma_f16(qf[0], kf[t * 2 + 0], st);
            st = wmma_f16(qf[1], kf[t * 2 + 1], st);
            s[t] = st;
        }

        // Issue V fragment loads now; they complete while softmax runs
        v16h vf[8];
#pragma unroll
        for (int kc = 0; kc < 2; ++kc)
#pragma unroll
            for (int t = 0; t < 4; ++t)
                vf[kc * 4 + t] = *(const v16h*)(vh + (size_t)(t * 16 + nn) * LL
                                                   + key0 + kc * 32 + g * 16);

        // Online softmax; write P (f16) to LDS in row-major [row][key] form
#pragma unroll
        for (int r = 0; r < 8; ++r) {
            float bm = fmaxf(fmaxf(s[0][r], s[1][r]), fmaxf(s[2][r], s[3][r]));
            bm = grp_max(bm);
            float nm   = fmaxf(rmax[r], bm);
            float corr = __expf(rmax[r] - nm);
            rmax[r] = nm;
            rsum[r] *= corr;
#pragma unroll
            for (int t = 0; t < 4; ++t) oacc[t][r] *= corr;
            float ps = 0.f;
#pragma unroll
            for (int t = 0; t < 4; ++t) {
                float p = __expf(s[t][r] - nm);
                pl[(r + 8 * g) * 64 + t * 16 + nn] = (_Float16)p;
                ps += p;
            }
            rsum[r] += grp_sum(ps);
        }
        __syncthreads();

        // O += P * V : A = P from LDS, B = V fragments already in registers
#pragma unroll
        for (int kc = 0; kc < 2; ++kc) {
            v16h pa = ld_a16(pl + nn * 64 + kc * 32 + g * 16);
#pragma unroll
            for (int t = 0; t < 4; ++t)
                oacc[t] = wmma_f16(pa, vf[kc * 4 + t], oacc[t]);
        }
        __syncthreads();
    }

    // Epilogue: normalize and store to (B, L, KD*H) f16, col = kd*H + h
#pragma unroll
    for (int r = 0; r < 8; ++r) {
        float inv = 1.0f / rsum[r];
        int row = qrow0 + r + 8 * g;
        size_t base = ((size_t)b * LL + row) * 512 + h;
#pragma unroll
        for (int t = 0; t < 4; ++t) {
            int kd = t * 16 + nn;
            o_s[base + (size_t)kd * HH] = (_Float16)(oacc[t][r] * inv);
        }
    }
}

// ---------- Host launch ----------
extern "C" void kernel_launch(void* const* d_in, const int* in_sizes, int n_in,
                              void* d_out, int out_size, void* d_ws, size_t ws_size,
                              hipStream_t stream) {
    (void)in_sizes; (void)n_in; (void)out_size; (void)ws_size;

    const float* query = (const float*)d_in[0];
    const float* keyi  = (const float*)d_in[1];
    const float* value = (const float*)d_in[2];
    const int*   wq_q  = (const int*)d_in[3];
    const int*   wk_q  = (const int*)d_in[4];
    const int*   wv_q  = (const int*)d_in[5];
    const int*   wo_q  = (const int*)d_in[6];
    const float* bq    = (const float*)d_in[7];
    const float* bk    = (const float*)d_in[8];
    const float* bv    = (const float*)d_in[9];
    const float* bo    = (const float*)d_in[10];
    const float* sc_q  = (const float*)d_in[11];
    const float* sc_k  = (const float*)d_in[12];
    const float* sc_v  = (const float*)d_in[13];
    const float* sc_o  = (const float*)d_in[14];
    const float* zp_q  = (const float*)d_in[15];
    const float* zp_k  = (const float*)d_in[16];
    const float* zp_v  = (const float*)d_in[17];
    const float* zp_o  = (const float*)d_in[18];

    const size_t NX = (size_t)BB * LL * DD;   // 4,194,304 activations
    const size_t NW = (size_t)DD * DD;        //   262,144 per weight

    _Float16* ws  = (_Float16*)d_ws;
    _Float16* xq  = ws;                 // f16 activations
    _Float16* xk  = xq  + NX;
    _Float16* xv  = xk  + NX;
    _Float16* wqt = xv  + NX;           // transposed dequantized weights
    _Float16* wkt = wqt + NW;
    _Float16* wvt = wkt + NW;
    _Float16* wot = wvt + NW;
    _Float16* qsm = wot + NW;           // q (B,H,L,KD)
    _Float16* ksm = qsm + NX;           // k (B,H,L,KD)
    _Float16* vsm = ksm + NX;           // v (B,H,KD,L)  transposed
    _Float16* osm = vsm + NX;           // attention out (B,L,KD*H)

    // 1) dequantize + transpose weights (all 512x512)
    dequant_transpose_k<<<1024, 256, 0, stream>>>(wq_q, wqt, sc_q, zp_q);
    dequant_transpose_k<<<1024, 256, 0, stream>>>(wk_q, wkt, sc_k, zp_k);
    dequant_transpose_k<<<1024, 256, 0, stream>>>(wv_q, wvt, sc_v, zp_v);
    dequant_transpose_k<<<1024, 256, 0, stream>>>(wo_q, wot, sc_o, zp_o);

    // 2) convert activations to f16
    f32_to_f16_k<<<(int)(NX / 256), 256, 0, stream>>>(query, xq, (int)NX);
    f32_to_f16_k<<<(int)(NX / 256), 256, 0, stream>>>(keyi,  xk, (int)NX);
    f32_to_f16_k<<<(int)(NX / 256), 256, 0, stream>>>(value, xv, (int)NX);

    // 3) QKV projections (WMMA GEMMs) with bias + layout shuffles
    gemm512_k<<<dim3(64, 8), 256, 0, stream>>>(xq, wqt, bq, qsm, 0);
    gemm512_k<<<dim3(64, 8), 256, 0, stream>>>(xk, wkt, bk, ksm, 0);
    gemm512_k<<<dim3(64, 8), 256, 0, stream>>>(xv, wvt, bv, vsm, 1);

    // 4) flash attention per (b,h): 32 heads x 32 row-blocks of 64
    attn_k<<<dim3(32, 32), 128, 0, stream>>>(qsm, ksm, vsm, osm);

    // 5) output projection -> f32 d_out with bias
    gemm512_k<<<dim3(64, 8), 256, 0, stream>>>(osm, wot, bo, d_out, 2);
}
